// pRecurrentLayer_76982993814373
// MI455X (gfx1250) — matrix operand order repaired and checked by reading, not compiled, not run on hardware
//
#include <hip/hip_runtime.h>

// ---------------- types ----------------
typedef __attribute__((ext_vector_type(16))) _Float16 v16h;
typedef __attribute__((ext_vector_type(8)))  float    v8f;
typedef __attribute__((ext_vector_type(4)))  unsigned int u32x4;

#define NB   64      // batch
#define NC   32      // channels in/out
#define NT   1024    // timesteps
#define R1   34      // theta1 rows
#define R2   1058    // theta2 rows
#define TC   128     // time chunk
#define NCH  (NT/TC) // 8 chunks
#define SNAP 16      // snapshot interval
#define NSNAP (NT/SNAP) // 64

// workspace byte offsets
#define OFF_A1     0u                 // 64*32*1024*4 = 8388608
#define OFF_STATES 8388608u           // 64*64*1024*4 = 16777216
#define OFF_W1G    25165824u          // [o][i] 32x32 f16 = 2048 B (pad to 4096)
#define OFF_WD     25169920u          // [o][c] 32x32 f16 = 2048 B
#define OFF_WG     25171968u          // [c][o][f] 32x32x32 f16 = 65536 B  (contiguous after WD)
#define OFF_B1     25237504u          // 128
#define OFF_B2     25237632u          // 128
#define OFF_FB     25237760u          // 4096

__device__ __forceinline__ float hw_tanh(float x) {
#if __has_builtin(__builtin_amdgcn_tanhf)
    return __builtin_amdgcn_tanhf(x);
#else
    float r;
    asm volatile("v_tanh_f32 %0, %1\n\tv_nop\n\tv_nop" : "=v"(r) : "v"(x));
    return r;
#endif
}

__device__ __forceinline__ float actf(float z) {
    return 0.134f + 0.962f * hw_tanh((z - 0.183f) * 24.10f);
}

__device__ __forceinline__ v8f wmma16(v16h a, v16h b, v8f c) {
    return __builtin_amdgcn_wmma_f32_16x16x32_f16(false, a, false, b, (short)0, c, false, false);
}

// A-fragment from a row-major [M][32] f16 matrix (rows 64B-aligned).
// Lane layout: M = mt*16 + (lane&15); K = kb..kb+15, kb = 16*(lane>=16).
__device__ __forceinline__ v16h afrag32(const _Float16* __restrict__ Arow, int hi) {
    const int kb = hi * 16;
    union { u32x4 q[2]; v16h v; } u;
    u.q[0] = *(const u32x4*)(Arow + kb);
    u.q[1] = *(const u32x4*)(Arow + 8 + kb);
    return u.v;
}

// B-fragment: tile stored [t][k], pitch 32 halfs; N = nt*16 + (lane&15); K = h + 16*hi.
__device__ __forceinline__ v16h bfrag(const _Float16* __restrict__ tile,
                                      int nt, int ln, int hi) {
    const u32x4* p = (const u32x4*)(tile + (nt * 16 + ln) * 32 + hi * 16);
    union { u32x4 q[2]; v16h v; } u;
    u.q[0] = p[0];
    u.q[1] = p[1];
    return u.v;
}

// ---------------- prep: normalized signed weights in WMMA-friendly layouts ----------------
__global__ __launch_bounds__(256) void prep_kernel(
    const float* __restrict__ th1, const float* __restrict__ th2,
    const float* __restrict__ beta, const float* __restrict__ cr,
    _Float16* __restrict__ W1g, _Float16* __restrict__ Wd, _Float16* __restrict__ Wg,
    float* __restrict__ b1, float* __restrict__ b2, float* __restrict__ fb) {
    __shared__ float red[256];
    const int blk = blockIdx.x, tid = threadIdx.x;
    const float* th = (blk < 32) ? th1 : th2;
    const int R = (blk < 32) ? R1 : R2;
    const int o = blk & 31;

    float partial = 0.f;
    for (int i = tid; i < R; i += 256) {
        float t = th[i * 32 + o];
        t = fminf(fmaxf(t, -10.0f), 10.0f);
        if (fabsf(t) < 0.01f) t = 0.f;
        partial += fabsf(t);
    }
    red[tid] = partial;
    __syncthreads();
    for (int s = 128; s > 0; s >>= 1) {
        if (tid < s) red[tid] += red[tid + s];
        __syncthreads();
    }
    const float inv = 1.0f / red[0];
    for (int i = tid; i < R; i += 256) {
        float t = th[i * 32 + o];
        t = fminf(fmaxf(t, -10.0f), 10.0f);
        if (fabsf(t) < 0.01f) t = 0.f;
        float w = t * inv;                 // signed normalized weight (pos/neg collapse)
        if (blk < 32) {
            if (i < 32) W1g[o * 32 + i] = (_Float16)w;
            else if (i == 32) b1[o] = w;   // ones-row -> bias
        } else {
            if (i < 1056) {
                int c = i / 33, r = i - c * 33;
                if (r == 0) Wd[o * 32 + c] = (_Float16)w;          // direct-a1 channel
                else        Wg[c * 1024 + o * 32 + (r - 1)] = (_Float16)w; // filtered
            } else if (i == 1056) b2[o] = w;
        }
    }
    if (blk == 0) {
        for (int idx = tid; idx < 1024; idx += 256) {
            float s = 1.0f / (1.0f + expf(-beta[idx]));
            fb[idx] = s * (1.0f + 0.5f * cr[idx]);
        }
    }
}

// ---------------- mac1: a1 = act(W1^T x + b1), per (b, chunk) ----------------
__global__ __launch_bounds__(256) void mac1_kernel(
    const float* __restrict__ x, const _Float16* __restrict__ W1g,
    const float* __restrict__ b1, float* __restrict__ a1) {
    extern __shared__ char smem[];
    _Float16* xT  = (_Float16*)smem;             // [128][32] f16
    _Float16* W1s = (_Float16*)(smem + 8192);    // [32][32]  f16
    const int tid = threadIdx.x;
    const int lane = tid & 31, wave = tid >> 5;
    const int hi = lane >> 4, ln = lane & 15;
    const int b = blockIdx.x >> 3, chunk = blockIdx.x & 7;
    const int t0 = chunk * TC;

    for (int idx = tid; idx < 128; idx += 256)   // 2048 B as uint4
        ((u32x4*)W1s)[idx] = ((const u32x4*)W1g)[idx];
    for (int idx = tid; idx < 32 * TC; idx += 256) {
        int c = idx >> 7, tl = idx & (TC - 1);
        xT[tl * 32 + c] = (_Float16)x[(b * NC + c) * NT + t0 + tl];
    }
    __syncthreads();

    v16h bf = bfrag(xT, wave, ln, hi);
    v16h a0 = afrag32(W1s + (0 * 16 + ln) * 32, hi);
    v16h a1f = afrag32(W1s + (1 * 16 + ln) * 32, hi);
    v8f acc0 = {}, acc1 = {};
    acc0 = wmma16(a0, bf, acc0);
    acc1 = wmma16(a1f, bf, acc1);

    const int tg = t0 + wave * 16 + ln;
#pragma unroll
    for (int r = 0; r < 8; ++r) {
        int o0 = r + hi * 8;
        a1[(b * NC + o0) * NT + tg] = actf(acc0[r] + b1[o0]);
        int o1 = 16 + r + hi * 8;
        a1[(b * NC + o1) * NT + tg] = actf(acc1[r] + b1[o1]);
    }
}

// ---------------- scan: EMA over T, snapshot every 16 steps ----------------
__global__ __launch_bounds__(1024) void scan_kernel(
    const float* __restrict__ a1, const float* __restrict__ fb,
    const float* __restrict__ m0, float* __restrict__ states) {
    extern __shared__ char smem[];
    float* a1s = (float*)smem;                  // [32][1024] f32, 128KB
    const int b = blockIdx.x, tid = threadIdx.x;
    for (int idx = tid; idx < NC * NT; idx += 1024) a1s[idx] = a1[b * NC * NT + idx];
    __syncthreads();
    const int c = tid >> 5;
    const float fbv = fb[tid];
    const float om = 1.0f - fbv;
    float m = m0[b * 1024 + tid];
    const float* row = a1s + c * NT;
    for (int sidx = 0; sidx < NSNAP; ++sidx) {
        states[(b * NSNAP + sidx) * 1024 + tid] = m;   // m_t (pre-update output)
        const float* rs = row + sidx * SNAP;
#pragma unroll
        for (int k = 0; k < SNAP; ++k) m = fbv * m + om * rs[k];
    }
}

// ---------------- mac2: fused filter regen + GEMM over 1056 channels + act ----------------
__global__ __launch_bounds__(256) void mac2_kernel(
    const float* __restrict__ a1, const _Float16* __restrict__ WdWg,
    const float* __restrict__ b2, const float* __restrict__ fb,
    const float* __restrict__ states, float* __restrict__ out) {
    extern __shared__ char smem[];
    _Float16* Wds = (_Float16*)smem;                    // [32][32]  f16 (direct term)
    _Float16* Wgs = (_Float16*)(smem + 2048);           // [32][32][32] f16 (filtered)
    float*    a1c = (float*)(smem + 67584);             // [32][128]  f32
    _Float16* a1T = (_Float16*)(smem + 67584 + 16384);  // [128][32]  f16
    _Float16* tF0 = (_Float16*)(smem + 67584 + 16384 + 8192);        // [128][32] f16
    _Float16* tF1 = (_Float16*)(smem + 67584 + 16384 + 8192 + 8192); // [128][32] f16

    const int tid = threadIdx.x;
    const int lane = tid & 31, wave = tid >> 5;
    const int hi = lane >> 4, ln = lane & 15;
    const int b = blockIdx.x >> 3, chunk = blockIdx.x & 7;
    const int t0 = chunk * TC;
    const int f = tid & 31, seg = tid >> 5;   // scan roles

    // stage Wd (2KB) + Wg (64KB), contiguous in ws: 4224 uint4
    for (int idx = tid; idx < 4224; idx += 256)
        ((u32x4*)Wds)[idx] = ((const u32x4*)WdWg)[idx];
    for (int idx = tid; idx < 32 * TC; idx += 256) {
        int c = idx >> 7, tl = idx & (TC - 1);
        float v = a1[(b * NC + c) * NT + t0 + tl];
        a1c[c * TC + tl] = v;
        a1T[tl * 32 + c] = (_Float16)v;
    }
    __syncthreads();

    v8f acc0 = {}, acc1 = {};
    {   // direct a1 channels: A[o][c] = Wd, B[c][t] = a1T
        v16h bf = bfrag(a1T, wave, ln, hi);
        v16h A0 = afrag32(Wds + (0 * 16 + ln) * 32, hi);
        v16h A1 = afrag32(Wds + (1 * 16 + ln) * 32, hi);
        acc0 = wmma16(A0, bf, acc0);
        acc1 = wmma16(A1, bf, acc1);
    }

    for (int c = 0; c < NC; ++c) {
        _Float16* tF = (c & 1) ? tF1 : tF0;   // double buffer
        // regenerate filtered[c, f, t0..t0+127] from snapshots (fully parallel)
        float fbv = fb[c * 32 + f];
        float om = 1.0f - fbv;
        float m = states[(b * NSNAP + chunk * 8 + seg) * 1024 + c * 32 + f];
        const float* arow = a1c + c * TC + seg * SNAP;
#pragma unroll
        for (int k = 0; k < SNAP; ++k) {
            tF[(seg * SNAP + k) * 32 + f] = (_Float16)m;
            m = fbv * m + om * arow[k];
        }
        __syncthreads();   // orders write(c) before read(c); read(c-2) before write(c)
        // accumulate: A[o][f] = Wg[c], B[f][t] = filtered tile
        v16h bf = bfrag(tF, wave, ln, hi);
        const _Float16* Ag = Wgs + c * 1024;
        v16h A0 = afrag32(Ag + (0 * 16 + ln) * 32, hi);
        v16h A1 = afrag32(Ag + (1 * 16 + ln) * 32, hi);
        acc0 = wmma16(A0, bf, acc0);
        acc1 = wmma16(A1, bf, acc1);
    }

    const int tg = t0 + wave * 16 + ln;
#pragma unroll
    for (int r = 0; r < 8; ++r) {
        int o0 = r + hi * 8;
        out[(b * NC + o0) * NT + tg] = actf(acc0[r] + b2[o0]);
        int o1 = 16 + r + hi * 8;
        out[(b * NC + o1) * NT + tg] = actf(acc1[r] + b2[o1]);
    }
}

extern "C" void kernel_launch(void* const* d_in, const int* in_sizes, int n_in,
                              void* d_out, int out_size, void* d_ws, size_t ws_size,
                              hipStream_t stream) {
    (void)in_sizes; (void)n_in; (void)out_size; (void)ws_size;
    const float* x    = (const float*)d_in[0];
    const float* th1  = (const float*)d_in[1];
    const float* th2  = (const float*)d_in[2];
    const float* beta = (const float*)d_in[3];
    const float* m0   = (const float*)d_in[4];
    const float* cr   = (const float*)d_in[5];
    float* out = (float*)d_out;
    char* ws = (char*)d_ws;

    float*    a1     = (float*)(ws + OFF_A1);
    float*    states = (float*)(ws + OFF_STATES);
    _Float16* W1g    = (_Float16*)(ws + OFF_W1G);
    _Float16* Wd     = (_Float16*)(ws + OFF_WD);
    _Float16* Wg     = (_Float16*)(ws + OFF_WG);
    float*    b1     = (float*)(ws + OFF_B1);
    float*    b2     = (float*)(ws + OFF_B2);
    float*    fb     = (float*)(ws + OFF_FB);

    prep_kernel<<<64, 256, 0, stream>>>(th1, th2, beta, cr, W1g, Wd, Wg, b1, b2, fb);
    mac1_kernel<<<NB * NCH, 256, 8192 + 2048, stream>>>(x, W1g, b1, a1);
    scan_kernel<<<NB, 1024, NC * NT * sizeof(float), stream>>>(a1, fb, m0, states);
    mac2_kernel<<<NB * NCH, 256, 67584 + 16384 + 8192 + 8192 + 8192, stream>>>(
        a1, Wd, b2, fb, states, out);
}
